// CRSDCell_45689862095274
// MI455X (gfx1250) — matrix operand order repaired and checked by reading, compiled, not run on hardware
//
#include <hip/hip_runtime.h>
#include <hip/hip_bf16.h>

typedef __attribute__((ext_vector_type(16))) _Float16 v16h;
typedef __attribute__((ext_vector_type(8)))  float    v8f;
typedef __attribute__((ext_vector_type(4)))  unsigned int v4u;
typedef __attribute__((ext_vector_type(8)))  int      v8i_t;
typedef __attribute__((ext_vector_type(4)))  int      v4i_t;

#define BSZ   2
#define TLEN  2048
#define DXC   512
#define DHC   512
#define NST   16
#define DKC   128
#define HALFC 64
#define ROWS  (BSZ*TLEN)   /* 4096 */
#define DN    (DHC*NST)    /* 8192 */
#define LN_EPS 1e-5f

#if __has_builtin(__builtin_amdgcn_tensor_load_to_lds)
#define HAVE_TDM 1
#else
#define HAVE_TDM 0
#endif

// ---- LDS byte-offset of a generic pointer to a __shared__ object ----------
__device__ __forceinline__ uint32_t lds_offset_of(const void* p) {
    return (uint32_t)(uintptr_t)(__attribute__((address_space(3))) const void*)p;
}

#if HAVE_TDM
// Issue a TDM 2-D tile load: tile (tileD1 rows x tileD0 elems) of an f16
// row-major tensor with row stride strideE (elements), into LDS at ldsoff,
// packed contiguously (row = tileD0 elems). D# built per CDNA5 ISA ch.8.
// 6-arg builtin form: (g0 v4u, g1 v8i, g2 v4i, g3 v4i, g4 v8i, cpol)
__device__ __forceinline__ void tdm_load_tile_f16(const _Float16* gtile, uint32_t ldsoff,
                                                  unsigned tileD0, unsigned tileD1,
                                                  unsigned strideE, unsigned dim1) {
    unsigned long long ga = (unsigned long long)(uintptr_t)gtile;
    v4u g0;
    g0.x = 1u;                                             // count=1, user mode
    g0.y = ldsoff;                                         // lds_addr (bytes)
    g0.z = (unsigned)(ga & 0xFFFFFFFFu);                   // global_addr[31:0]
    g0.w = (unsigned)((ga >> 32) & 0x01FFFFFFu) | (2u << 30);  // addr[56:32] | type=2
    v8i_t g1;
    g1[0] = (int)(1u << 16);                               // data_size=1 (2 bytes)
    g1[1] = (int)((strideE & 0xFFFFu) << 16);              // tensor_dim0[15:0]
    g1[2] = (int)(((strideE >> 16) & 0xFFFFu) | ((dim1 & 0xFFFFu) << 16));
    g1[3] = (int)(((dim1 >> 16) & 0xFFFFu) | (tileD0 << 16));   // tile_dim0
    g1[4] = (int)tileD1;                                   // tile_dim1 | tile_dim2=0
    g1[5] = (int)strideE;                                  // tensor_dim0_stride lo
    g1[6] = 0;                                             // stride hi | dim1_stride lo
    g1[7] = 0;
    v4i_t z4 = {0, 0, 0, 0};
    v8i_t z8 = {0, 0, 0, 0, 0, 0, 0, 0};
    __builtin_amdgcn_tensor_load_to_lds(g0, g1, z4, z4, z8, 0);
}
__device__ __forceinline__ void tile_wait() {
    __builtin_amdgcn_s_wait_tensorcnt(0);
}
#else
// Fallback: per-lane async global->LDS 16-byte copies (ASYNCcnt path).
__device__ __forceinline__ void async_g2l_b128(uint32_t lds_byte_off, const void* gaddr) {
    asm volatile("global_load_async_to_lds_b128 %0, %1, off"
                 :: "v"(lds_byte_off), "v"((unsigned long long)(uintptr_t)gaddr)
                 : "memory");
}
__device__ __forceinline__ void tile_wait() {
#if __has_builtin(__builtin_amdgcn_s_wait_asynccnt)
    __builtin_amdgcn_s_wait_asynccnt(0);
#else
    asm volatile("s_wait_asynccnt 0x0" ::: "memory");
#endif
}
#endif

// ---------------------------------------------------------------- utilities
__global__ __launch_bounds__(256) void cvt_f32_f16_kernel(const float* __restrict__ src,
                                                          _Float16* __restrict__ dst, int n) {
    int i = blockIdx.x * 256 + threadIdx.x;
    if (i < n) dst[i] = (_Float16)src[i];
}

// ------------------------------------------------- LayerNorm + RoPE + delta
__global__ __launch_bounds__(256) void ln_rope_delta_kernel(
    const float* __restrict__ x_seq, const float* __restrict__ sinp,
    const float* __restrict__ cosp, const float* __restrict__ ln_g,
    const float* __restrict__ ln_b, const float* __restrict__ Wdelta,
    const float* __restrict__ bdelta, _Float16* __restrict__ xssm16,
    _Float16* __restrict__ xseq16, float* __restrict__ delta_out) {
    __shared__ float sv[DXC];
    __shared__ float red[256];
    int row = blockIdx.x;          // 0..4095
    int t   = row & (TLEN - 1);
    int tid = threadIdx.x;
    const float* xr = x_seq + (size_t)row * DXC;
    float x0 = xr[tid], x1 = xr[tid + 256];
    xseq16[(size_t)row * DXC + tid]       = (_Float16)x0;
    xseq16[(size_t)row * DXC + tid + 256] = (_Float16)x1;

    red[tid] = x0 + x1;
    __syncthreads();
    for (int s = 128; s > 0; s >>= 1) { if (tid < s) red[tid] += red[tid + s]; __syncthreads(); }
    float mu = red[0] * (1.0f / DXC);
    __syncthreads();
    float d0 = x0 - mu, d1 = x1 - mu;
    red[tid] = d0 * d0 + d1 * d1;
    __syncthreads();
    for (int s = 128; s > 0; s >>= 1) { if (tid < s) red[tid] += red[tid + s]; __syncthreads(); }
    float rs = rsqrtf(red[0] * (1.0f / DXC) + LN_EPS);
    __syncthreads();

    sv[tid]       = d0 * rs * ln_g[tid]       + ln_b[tid];
    sv[tid + 256] = d1 * rs * ln_g[tid + 256] + ln_b[tid + 256];
    __syncthreads();

    const float* sr = sinp + (size_t)t * HALFC;
    const float* cr = cosp + (size_t)t * HALFC;
    float r0;
    {
        int e = tid;
        if (e < HALFC)            r0 = sv[e] * cr[e] - sv[e + HALFC] * sr[e];
        else if (e < 2 * HALFC) { int i = e - HALFC; r0 = sv[i] * sr[i] + sv[e] * cr[i]; }
        else                      r0 = sv[e];
    }
    float r1 = sv[tid + 256];  // rope only touches first 128 dims
    xssm16[(size_t)row * DXC + tid]       = (_Float16)r0;
    xssm16[(size_t)row * DXC + tid + 256] = (_Float16)r1;

    red[tid] = r0 * Wdelta[tid] + r1 * Wdelta[tid + 256];
    __syncthreads();
    for (int s = 128; s > 0; s >>= 1) { if (tid < s) red[tid] += red[tid + s]; __syncthreads(); }
    if (tid == 0) {
        float z  = red[0] + bdelta[0];
        float sp = (z > 20.0f) ? z : log1pf(__expf(z));
        delta_out[row] = fminf(sp, 0.5f);
    }
}

// --------------------------------------------------------- WMMA f16 GEMM
// out[M,N] = act(A[M,K] @ W[K,N] + bias), A/W f16, out f32 (+ optional f16).
// 128x128 tile per 256-thread block (8 waves), BK = 32.
// launch_bounds(256,1): allow full register budget so all 8 B-fragments
// stay live -> single dscnt wait + 8 back-to-back WMMAs per k-step.
__global__ __launch_bounds__(256, 1) void wmma_gemm_f16_kernel(
    const _Float16* __restrict__ A, const _Float16* __restrict__ W,
    const float* __restrict__ bias, float* __restrict__ out,
    _Float16* __restrict__ out16, int M, int N, int K, int act) {
    __shared__ __align__(16) _Float16 sA[128 * 32];    // [m][k]
    __shared__ __align__(16) _Float16 sBT[128 * 32];   // [n][k] (transposed)
    int tid  = threadIdx.x;
    int wave = tid >> 5;
    int lane = tid & 31;
    int lm   = lane & 15;
    int g    = lane >> 4;
    int m0   = blockIdx.y * 128;
    int n0   = blockIdx.x * 128;

    v8f acc[8] = {};

    for (int kk = 0; kk < K; kk += 32) {
        const _Float16* Ag = A + (size_t)m0 * K + kk;
#if HAVE_TDM
        if (tid == 0) {
            tdm_load_tile_f16(Ag, lds_offset_of(&sA[0]), 32u, 128u,
                              (unsigned)K, (unsigned)M);
        }
#else
        {   // 512 x 16B chunks, 2 per thread
            #pragma unroll
            for (int j = 0; j < 2; j++) {
                int c  = tid * 2 + j;
                int r  = c >> 2;
                int cc = (c & 3) * 8;
                async_g2l_b128(lds_offset_of(&sA[r * 32 + cc]),
                               Ag + (size_t)r * K + cc);
            }
        }
#endif
        // stage W tile transposed (sync)
        const _Float16* Wg = W + (size_t)kk * N + n0;
        #pragma unroll
        for (int j = 0; j < 2; j++) {
            int c  = tid * 2 + j;
            int kr = c >> 4;
            int nc = (c & 15) * 8;
            union { uint4 u; _Float16 h[8]; } tmp;
            tmp.u = *(const uint4*)(Wg + (size_t)kr * N + nc);
            #pragma unroll
            for (int i = 0; i < 8; i++) sBT[(nc + i) * 32 + kr] = tmp.h[i];
        }
        if (kk + 32 < K) {  // prefetch next W tile (global_prefetch_b8)
            __builtin_prefetch(W + (size_t)(kk + 32 + (tid & 31)) * N + n0, 0, 1);
        }
        tile_wait();          // TENSORcnt / ASYNCcnt drain for A tile
        __syncthreads();

        // A fragment: 16x32 f16 per ISA layout
        int am = wave * 16 + lm;
        v16h afrag;
        *(uint4*)(&afrag)       = *(const uint4*)(&sA[am * 32 + 8 * g]);
        *((uint4*)(&afrag) + 1) = *(const uint4*)(&sA[am * 32 + 16 + 8 * g]);

        // register-block all 8 B fragments, then back-to-back WMMAs
        v16h bf[8];
        #pragma unroll
        for (int nt = 0; nt < 8; nt++) {
            int bn = nt * 16 + lm;
            *(uint4*)(&bf[nt])       = *(const uint4*)(&sBT[bn * 32 + 8 * g]);
            *((uint4*)(&bf[nt]) + 1) = *(const uint4*)(&sBT[bn * 32 + 16 + 8 * g]);
        }
        // first WMMA: no reuse; remaining 7: reuse_a hint (same A operand)
        acc[0] = __builtin_amdgcn_wmma_f32_16x16x32_f16(
            false, afrag, false, bf[0], (short)0, acc[0], false, false);
        #pragma unroll
        for (int nt = 1; nt < 8; nt++) {
            acc[nt] = __builtin_amdgcn_wmma_f32_16x16x32_f16(
                false, afrag, false, bf[nt], (short)0, acc[nt], true, false);
        }
        __syncthreads();
    }

    #pragma unroll
    for (int nt = 0; nt < 8; nt++) {
        int col = n0 + nt * 16 + lm;
        float bv = bias ? bias[col] : 0.0f;
        #pragma unroll
        for (int r = 0; r < 8; r++) {
            int row = m0 + wave * 16 + r + 8 * g;
            float v = acc[nt][r] + bv;
            if (act == 1) v = v / (1.0f + __expf(-v));  // silu
            size_t o = (size_t)row * N + col;
            out[o] = v;
            if (out16) out16[o] = (_Float16)v;
        }
    }
}

// -------------------------------------------- column mean of B_cont (2-stage)
__global__ __launch_bounds__(256) void colpart_kernel(const float* __restrict__ Bc,
                                                      float* __restrict__ colpart) {
    int j     = blockIdx.x * 256 + threadIdx.x;  // column 0..8191
    int slice = blockIdx.y;                      // 0..7
    int r0    = slice * (ROWS / 8);
    float s = 0.0f;
    for (int r = r0; r < r0 + ROWS / 8; r++) s += Bc[(size_t)r * DN + j];
    colpart[(size_t)slice * DN + j] = s;
}
__global__ __launch_bounds__(256) void colmean_fin_kernel(const float* __restrict__ colpart,
                                                          float* __restrict__ colmean) {
    int j = blockIdx.x * 256 + threadIdx.x;
    float s = 0.0f;
    #pragma unroll
    for (int p = 0; p < 8; p++) s += colpart[(size_t)p * DN + j];
    colmean[j] = s * (1.0f / ROWS);
}

// ------------------------------------------------------- sequential scan
__global__ __launch_bounds__(256) void scan_kernel(
    const float* __restrict__ Bc_raw, const float* __restrict__ colmean,
    const float* __restrict__ delta, const float* __restrict__ A_log,
    float* __restrict__ h_seq) {
    int j = blockIdx.x * 256 + threadIdx.x;   // 0..16383 : b*8192 + (d*16+n)
    int b = j >> 13;
    int c = j & (DN - 1);
    float Ac   = -__expf(A_log[c]);
    float mean = colmean[c];
    const float* br = Bc_raw + (size_t)b * TLEN * DN + c;
    float*       ho = h_seq  + (size_t)b * TLEN * DN + c;
    const float* dl = delta + b * TLEN;
    float h = 0.0f, prevA = 1.0f;
    bool  bigA = fabsf(Ac) > 1e-4f;
    for (int t = 0; t < TLEN; t++) {
        float de  = dl[t];
        float dA  = de * Ac;
        float eA  = __expf(dA);
        float Abr = fminf(fmaxf(eA, 0.0f), 1.0f);
        float inj = bigA ? (expm1f(dA) / Ac) : de;
        float Bcv = 0.5f * (br[(size_t)t * DN] - mean);
        float Bb  = inj * Bcv;
        if (!isfinite(Bb)) Bb = (Bb != Bb) ? 0.0f : (Bb > 0.0f ? 1000.0f : -1000.0f);
        float a = fmaxf(prevA, 1e-6f);        // A_shift: 1 at t=0, A_bar[t-1] after
        h = h / a + Bb;                        // == cumsum(Bb*pp)/pp of reference
        ho[(size_t)t * DN] = h;
        prevA = Abr;
    }
}

// ------------------------------------------ h_out = sum_n h_seq * C_coeff
__global__ __launch_bounds__(256) void einsum_kernel(const float* __restrict__ hseq,
                                                     const float* __restrict__ Cc,
                                                     float* __restrict__ hout) {
    int i = blockIdx.x * 256 + threadIdx.x;   // 0 .. ROWS*DHC-1
    const float4* h4 = (const float4*)(hseq + (size_t)i * NST);
    const float4* c4 = (const float4*)(Cc   + (size_t)i * NST);
    float s = 0.0f;
    #pragma unroll
    for (int q = 0; q < 4; q++) {
        float4 a = h4[q], b = c4[q];
        s += a.x * b.x + a.y * b.y + a.z * b.z + a.w * b.w;
    }
    hout[i] = s;
}

// -------------------------------------- y_ssm = h_out + D_scale*(xWD+bD)
__global__ __launch_bounds__(256) void yssm_kernel(const float* __restrict__ hout,
                                                   const float* __restrict__ xWD,
                                                   const float* __restrict__ D_scale,
                                                   float* __restrict__ yssm,
                                                   _Float16* __restrict__ yssm16) {
    int i = blockIdx.x * 256 + threadIdx.x;   // ROWS*DHC
    int d = i & (DHC - 1);
    float v = hout[i] + D_scale[d] * xWD[i];
    yssm[i]   = v;
    yssm16[i] = (_Float16)v;
}

// ---------------------------------------------------------- RoPE on k
__global__ __launch_bounds__(256) void ropek_kernel(const float* __restrict__ klin,
                                                    const float* __restrict__ sinp,
                                                    const float* __restrict__ cosp,
                                                    float* __restrict__ kout,
                                                    _Float16* __restrict__ k16) {
    int i   = blockIdx.x * 256 + threadIdx.x;  // ROWS*HALFC
    int row = i >> 6;
    int j   = i & (HALFC - 1);
    int t   = row & (TLEN - 1);
    float s  = sinp[(size_t)t * HALFC + j];
    float c  = cosp[(size_t)t * HALFC + j];
    float x1 = klin[(size_t)row * DKC + j];
    float x2 = klin[(size_t)row * DKC + HALFC + j];
    float o1 = x1 * c - x2 * s;
    float o2 = x1 * s + x2 * c;
    kout[(size_t)row * DKC + j]          = o1;
    kout[(size_t)row * DKC + HALFC + j]  = o2;
    k16 [(size_t)row * DKC + j]          = (_Float16)o1;
    k16 [(size_t)row * DKC + HALFC + j]  = (_Float16)o2;
}

// ----------------------------------------------------- y = y_ssm + h_mem
__global__ __launch_bounds__(256) void final_add_kernel(const float* __restrict__ yssm,
                                                        const float* __restrict__ hmem,
                                                        float* __restrict__ yout) {
    int i = blockIdx.x * 256 + threadIdx.x;
    yout[i] = yssm[i] + hmem[i];
}

// --------------------------------------------------------------- launcher
extern "C" void kernel_launch(void* const* d_in, const int* in_sizes, int n_in,
                              void* d_out, int out_size, void* d_ws, size_t ws_size,
                              hipStream_t stream) {
    const float* x_seq   = (const float*)d_in[0];
    const float* sinp    = (const float*)d_in[1];
    const float* cosp    = (const float*)d_in[2];
    const float* A_log   = (const float*)d_in[3];
    const float* ln_g    = (const float*)d_in[4];
    const float* ln_b    = (const float*)d_in[5];
    const float* Wdelta  = (const float*)d_in[6];
    const float* bdelta  = (const float*)d_in[7];
    const float* WB      = (const float*)d_in[8];
    const float* bB      = (const float*)d_in[9];
    const float* WC      = (const float*)d_in[10];
    const float* bC      = (const float*)d_in[11];
    const float* WD      = (const float*)d_in[12];
    const float* bD      = (const float*)d_in[13];
    const float* D_scale = (const float*)d_in[14];
    const float* Wk      = (const float*)d_in[15];
    const float* bk      = (const float*)d_in[16];
    const float* Wv      = (const float*)d_in[17];
    const float* bv      = (const float*)d_in[18];
    const float* Wr      = (const float*)d_in[19];
    const float* br      = (const float*)d_in[20];
    const float* Mm      = (const float*)d_in[21];

    // d_out layout: y | h_seq | k | v (flat, return order)
    float* y_out    = (float*)d_out;
    float* hseq_out = y_out + (size_t)ROWS * DHC;
    float* k_out    = hseq_out + (size_t)ROWS * DN;
    float* v_out    = k_out + (size_t)ROWS * DKC;

    size_t off = 0;
    char* base = (char*)d_ws;
    auto take = [&](size_t bytes) -> void* {
        void* p = base + off;
        off += (bytes + 255) & ~(size_t)255;
        return p;
    };
    _Float16* xssm16  = (_Float16*)take((size_t)ROWS * DXC * 2);
    _Float16* xseq16  = (_Float16*)take((size_t)ROWS * DXC * 2);
    float*    delta   = (float*)take((size_t)ROWS * 4);
    _Float16* WB16    = (_Float16*)take((size_t)DXC * DN * 2);
    _Float16* WC16    = (_Float16*)take((size_t)DXC * DN * 2);
    _Float16* WD16    = (_Float16*)take((size_t)DXC * DHC * 2);
    _Float16* Wk16    = (_Float16*)take((size_t)DHC * DKC * 2);
    _Float16* Wv16    = (_Float16*)take((size_t)DHC * DKC * 2);
    _Float16* Wr16    = (_Float16*)take((size_t)DKC * DHC * 2);
    _Float16* M16     = (_Float16*)take((size_t)DKC * DKC * 2);
    float*    Bc_raw  = (float*)take((size_t)ROWS * DN * 4);
    float*    Cc      = (float*)take((size_t)ROWS * DN * 4);
    float*    colpart = (float*)take((size_t)8 * DN * 4);
    float*    colmean = (float*)take((size_t)DN * 4);
    float*    h_out   = (float*)take((size_t)ROWS * DHC * 4);
    float*    xWD     = (float*)take((size_t)ROWS * DHC * 4);
    float*    yssm    = (float*)take((size_t)ROWS * DHC * 4);
    _Float16* yssm16  = (_Float16*)take((size_t)ROWS * DHC * 2);
    float*    klin    = (float*)take((size_t)ROWS * DKC * 4);
    _Float16* k16     = (_Float16*)take((size_t)ROWS * DKC * 2);
    float*    vhat    = (float*)take((size_t)ROWS * DKC * 4);
    _Float16* vhat16  = (_Float16*)take((size_t)ROWS * DKC * 2);
    float*    hmem    = (float*)take((size_t)ROWS * DHC * 4);

    auto cvt = [&](const float* s, _Float16* d, int n) {
        cvt_f32_f16_kernel<<<(n + 255) / 256, 256, 0, stream>>>(s, d, n);
    };
    cvt(WB, WB16, DXC * DN);
    cvt(WC, WC16, DXC * DN);
    cvt(WD, WD16, DXC * DHC);
    cvt(Wk, Wk16, DHC * DKC);
    cvt(Wv, Wv16, DHC * DKC);
    cvt(Wr, Wr16, DKC * DHC);
    cvt(Mm, M16,  DKC * DKC);

    ln_rope_delta_kernel<<<ROWS, 256, 0, stream>>>(
        x_seq, sinp, cosp, ln_g, ln_b, Wdelta, bdelta, xssm16, xseq16, delta);

    wmma_gemm_f16_kernel<<<dim3(DN / 128, ROWS / 128), 256, 0, stream>>>(
        xssm16, WB16, bB, Bc_raw, (_Float16*)nullptr, ROWS, DN, DXC, 1);
    wmma_gemm_f16_kernel<<<dim3(DN / 128, ROWS / 128), 256, 0, stream>>>(
        xssm16, WC16, bC, Cc, (_Float16*)nullptr, ROWS, DN, DXC, 1);
    wmma_gemm_f16_kernel<<<dim3(DHC / 128, ROWS / 128), 256, 0, stream>>>(
        xseq16, WD16, bD, xWD, (_Float16*)nullptr, ROWS, DHC, DXC, 0);

    colpart_kernel<<<dim3(DN / 256, 8), 256, 0, stream>>>(Bc_raw, colpart);
    colmean_fin_kernel<<<DN / 256, 256, 0, stream>>>(colpart, colmean);

    scan_kernel<<<(BSZ * DN) / 256, 256, 0, stream>>>(Bc_raw, colmean, delta, A_log, hseq_out);

    einsum_kernel<<<(ROWS * DHC) / 256, 256, 0, stream>>>(hseq_out, Cc, h_out);

    yssm_kernel<<<(ROWS * DHC) / 256, 256, 0, stream>>>(h_out, xWD, D_scale, yssm, yssm16);

    wmma_gemm_f16_kernel<<<dim3(DKC / 128, ROWS / 128), 256, 0, stream>>>(
        yssm16, Wk16, bk, klin, (_Float16*)nullptr, ROWS, DKC, DHC, 0);
    wmma_gemm_f16_kernel<<<dim3(DKC / 128, ROWS / 128), 256, 0, stream>>>(
        yssm16, Wv16, bv, v_out, (_Float16*)nullptr, ROWS, DKC, DHC, 0);

    ropek_kernel<<<(ROWS * HALFC) / 256, 256, 0, stream>>>(klin, sinp, cosp, k_out, k16);

    wmma_gemm_f16_kernel<<<dim3(DKC / 128, ROWS / 128), 256, 0, stream>>>(
        k16, M16, (const float*)nullptr, vhat, vhat16, ROWS, DKC, DKC, 0);
    wmma_gemm_f16_kernel<<<dim3(DHC / 128, ROWS / 128), 256, 0, stream>>>(
        vhat16, Wr16, br, hmem, (_Float16*)nullptr, ROWS, DHC, DKC, 0);

    final_add_kernel<<<(ROWS * DHC) / 256, 256, 0, stream>>>(yssm, hmem, y_out);
}
